// PPONet_78537771975226
// MI455X (gfx1250) — matrix operand order repaired and checked by reading, not compiled, not run on hardware
//
#include <hip/hip_runtime.h>

#define BATCH 8
#define NN 512
#define EMB 64
#define TITER 5
#define MT 2          // m-tiles (16 rows each) per wave

typedef __attribute__((ext_vector_type(16))) __bf16         v16bf;
typedef __attribute__((ext_vector_type(8)))  float          v8f;
typedef __attribute__((ext_vector_type(8)))  unsigned short u16x8;

__device__ __forceinline__ __bf16 us_as_bf(unsigned short u) {
  union { unsigned short u; __bf16 b; } c; c.u = u; return c.b;
}

__device__ __forceinline__ unsigned short f2bf(float f) {
  unsigned int u = __float_as_uint(f);
  unsigned int r = u + 0x7FFFu + ((u >> 16) & 1u);   // round-to-nearest-even
  return (unsigned short)(r >> 16);
}

// Async global->LDS copy, 16 bytes per active lane (GLOBAL_LOAD_ASYNC_TO_LDS_B128,
// GV mode, tracked by ASYNCcnt).
__device__ __forceinline__ void async_copy_b128(unsigned ldsoff, unsigned long long gaddr) {
  asm volatile("global_load_async_to_lds_b128 %0, %1, off"
               :: "v"(ldsoff), "v"(gaddr) : "memory");
}
__device__ __forceinline__ void wait_asynccnt0() {
  asm volatile("s_wait_asynccnt 0" ::: "memory");
}

// A-fragment (16x32 bf16, row per lane&15): elems 0..7 at p[0..7], elems 8..15 at p[16..23]
__device__ __forceinline__ v16bf load_a_frag(const unsigned short* p) {
  u16x8 lo = *(const u16x8*)p;
  u16x8 hi = *(const u16x8*)(p + 16);
  v16bf v;
#pragma unroll
  for (int i = 0; i < 8; ++i) { v[i] = us_as_bf(lo[i]); v[i + 8] = us_as_bf(hi[i]); }
  return v;
}

// B-fragment (32x16 bf16, column per lane&15): elems 0..7 at p[0..7], elems 8..15 at p[16..23]
__device__ __forceinline__ v16bf load_b_frag(const unsigned short* p) {
  u16x8 lo = *(const u16x8*)p;
  u16x8 hi = *(const u16x8*)(p + 16);
  v16bf v;
#pragma unroll
  for (int i = 0; i < 8; ++i) { v[i] = us_as_bf(lo[i]); v[i + 8] = us_as_bf(hi[i]); }
  return v;
}

// ---------------- prep kernels ----------------
__global__ void k_cvt_ws(const float* __restrict__ Ws, unsigned short* __restrict__ WsBf, int total) {
  for (int i = blockIdx.x * blockDim.x + threadIdx.x; i < total; i += gridDim.x * blockDim.x)
    WsBf[i] = f2bf(Ws[i]);
}

__global__ void k_w2t(const float* __restrict__ W2, unsigned short* __restrict__ W2T) {
  int i = blockIdx.x * blockDim.x + threadIdx.x;   // 4096 threads
  int k = i >> 6, n = i & 63;
  W2T[n * EMB + k] = f2bf(W2[k * EMB + n]);
}

__global__ void k_zero16(unsigned short* __restrict__ p, int total) {
  for (int i = blockIdx.x * blockDim.x + threadIdx.x; i < total; i += gridDim.x * blockDim.x)
    p[i] = 0;
}

// ---------------- iteration-invariant terms: s13 = s1 + s3 + b2 ----------------
__global__ __launch_bounds__(64) void k_s13(
    const float* __restrict__ xv, const float* __restrict__ Ws,
    const float* __restrict__ W1a, const float* __restrict__ b1a,
    const float* __restrict__ W1b, const float* __restrict__ b1b,
    const float* __restrict__ b2,
    const float* __restrict__ W3,  const float* __restrict__ b3,
    const float* __restrict__ w4,  const float* __restrict__ b4,
    float* __restrict__ s13) {
  __shared__ float xl[8];
  __shared__ float h[EMB];
  __shared__ float s3p[EMB];
  const int bn = blockIdx.x;          // b*512 + n
  const int b  = bn >> 9;
  const int n  = bn & (NN - 1);
  const int e  = threadIdx.x;

  if (e < 5) xl[e] = xv[bn * 5 + e];
  __syncthreads();

  float hv = b1a[e];
#pragma unroll
  for (int d = 0; d < 5; ++d) hv += xl[d] * W1a[d * EMB + e];
  h[e] = hv > 0.f ? hv : 0.f;
  __syncthreads();

  float s1 = b1b[e];
  for (int k = 0; k < EMB; ++k) s1 += h[k] * W1b[k * EMB + e];

  const float w4e = w4[e], b4e = b4[e];
  float sp = 0.f;
  const float* wcol = Ws + (size_t)b * NN * NN + n;
  for (int m = 0; m < NN; ++m) {
    float t = wcol[(size_t)m * NN] * w4e + b4e;
    sp += t > 0.f ? t : 0.f;
  }
  s3p[e] = sp;
  __syncthreads();

  float s3 = b3[e];
  for (int k = 0; k < EMB; ++k) s3 += s3p[k] * W3[k * EMB + e];

  s13[(size_t)bn * EMB + e] = s1 + s3 + b2[e];
}

// ---------------- WMMA message-passing iteration ----------------
// Block = 4 waves, one batch, 8 consecutive 16-row m-tiles (2 per wave).
// The 64x32 bf16 B panel (mu^T slice) is shared by all waves -> double-buffered
// in LDS via async global->LDS copies overlapping the WMMAs of the current step.
#define BSTRIDE 40   // padded LDS row stride (ushorts), 16B-aligned

__global__ __launch_bounds__(128) void k_iter(
    const unsigned short* __restrict__ WsBf,   // [B][512][512] bf16
    const unsigned short* __restrict__ muT,    // [B][64][512]  bf16 (transposed mu)
    const unsigned short* __restrict__ W2T,    // [64][64]      bf16 (transposed W2)
    const float* __restrict__ s13,             // [B][512][64]
    float* __restrict__ mu_out,                // [B][512][64]
    unsigned short* __restrict__ muT_out) {    // [B][64][512]
  __shared__ __align__(16) unsigned short bbuf[2][EMB * BSTRIDE]; // double-buffered B panel
  __shared__ __align__(16) unsigned short plds[4 * MT * 16 * 72]; // per-wave P tiles

  const int wib  = threadIdx.x >> 5;
  const int b    = blockIdx.x >> 2;                  // 4 blocks per batch
  const int m0   = (((blockIdx.x & 3) << 2) + wib) * (MT * 16);
  const int lane = threadIdx.x & 31;
  const int l15  = lane & 15;
  const int half = lane >> 4;
  const int ko   = half << 3;

  // per-thread staging assignment: 2 x 16B chunks of the 4KB panel, per buffer
  const int c0   = threadIdx.x * 2;
  const int row0 = c0 >> 2,       kof0 = (c0 & 3) * 8;
  const int row1 = (c0 + 1) >> 2, kof1 = ((c0 + 1) & 3) * 8;
  unsigned lo0[2], lo1[2];
#pragma unroll
  for (int q = 0; q < 2; ++q) {
    lo0[q] = (unsigned)(size_t)(&bbuf[q][0] + row0 * BSTRIDE + kof0);
    lo1[q] = (unsigned)(size_t)(&bbuf[q][0] + row1 * BSTRIDE + kof1);
  }
  const unsigned short* gB0 = muT + (size_t)(b * EMB + row0) * NN + kof0;
  const unsigned short* gB1 = muT + (size_t)(b * EMB + row1) * NN + kof1;

  v8f acc[MT][4] = {};
  const unsigned short* arow[MT];
#pragma unroll
  for (int t = 0; t < MT; ++t)
    arow[t] = WsBf + (size_t)(b * NN + m0 + t * 16 + l15) * NN;

  // prologue: stage panel for step 0
  async_copy_b128(lo0[0], (unsigned long long)gB0);
  async_copy_b128(lo1[0], (unsigned long long)gB1);

  // Phase 1: P = Ws @ mu   (16 k-steps x MT m-tiles x 4 e-tiles)
  for (int i = 0; i < NN / 32; ++i) {
    const int k0 = i * 32;
    wait_asynccnt0();            // this wave's panel copies for step i landed
    __syncthreads();             // everyone's landed; step i-1 consumers done
    if (i + 1 < NN / 32) {       // prefetch step i+1 into the other buffer
      const int kn = k0 + 32;
      const int nb = (i + 1) & 1;
      async_copy_b128(lo0[nb], (unsigned long long)(gB0 + kn));
      async_copy_b128(lo1[nb], (unsigned long long)(gB1 + kn));
    }
    const unsigned short* bb = &bbuf[i & 1][0];
    v16bf A[MT];
#pragma unroll
    for (int t = 0; t < MT; ++t) A[t] = load_a_frag(arow[t] + k0 + ko);
#pragma unroll
    for (int j = 0; j < 4; ++j) {
      v16bf Bv = load_b_frag(bb + (j * 16 + l15) * BSTRIDE + half * 8);
#pragma unroll
      for (int t = 0; t < MT; ++t)
        acc[t][j] = __builtin_amdgcn_wmma_f32_16x16x32_bf16(false, A[t], false, Bv,
                                                            (short)0, acc[t][j], false, false);
    }
  }

  // Spill P tiles to LDS as bf16 (C-layout -> row-major)
#pragma unroll
  for (int t = 0; t < MT; ++t) {
    unsigned short* my = plds + (wib * MT + t) * (16 * 72);
#pragma unroll
    for (int j = 0; j < 4; ++j) {
#pragma unroll
      for (int r = 0; r < 8; ++r) {
        int row = r + half * 8;
        int col = j * 16 + l15;
        my[row * 72 + col] = f2bf(acc[t][j][r]);
      }
    }
  }
  __syncthreads();

  // Phase 2: s2 = P @ W2   (MT m-tiles x 2 k-steps x 4 e-tiles)
  v8f acc2[MT][4] = {};
#pragma unroll
  for (int t = 0; t < MT; ++t) {
    const unsigned short* my = plds + (wib * MT + t) * (16 * 72);
#pragma unroll
    for (int k0 = 0; k0 < EMB; k0 += 32) {
      v16bf A2 = load_a_frag(my + l15 * 72 + k0 + ko);
#pragma unroll
      for (int j = 0; j < 4; ++j) {
        const unsigned short* bp = W2T + (j * 16 + l15) * EMB + k0 + half * 8;
        v16bf Bv = load_b_frag(bp);
        acc2[t][j] = __builtin_amdgcn_wmma_f32_16x16x32_bf16(false, A2, false, Bv,
                                                             (short)0, acc2[t][j], false, false);
      }
    }
  }

  // Epilogue: mu = relu(s13 + s2); write fp32 mu + bf16 mu^T for the next iteration
#pragma unroll
  for (int t = 0; t < MT; ++t) {
#pragma unroll
    for (int j = 0; j < 4; ++j) {
#pragma unroll
      for (int r = 0; r < 8; ++r) {
        int m = m0 + t * 16 + r + half * 8;
        int e = j * 16 + l15;
        size_t idx = (size_t)(b * NN + m) * EMB + e;
        float v = acc2[t][j][r] + s13[idx];
        v = v > 0.f ? v : 0.f;
        mu_out[idx] = v;
        muT_out[(size_t)(b * EMB + e) * NN + m] = f2bf(v);
      }
    }
  }
}

// ---------------- readout head ----------------
__global__ __launch_bounds__(64) void k_batch(
    const float* __restrict__ mu,
    const float* __restrict__ W6,  const float* __restrict__ b6,
    const float* __restrict__ W51, const float* __restrict__ b51,
    float* __restrict__ gpart) {               // [B][64]
  __shared__ float cs[EMB];
  __shared__ float rg[EMB];
  const int b = blockIdx.x;
  const int e = threadIdx.x;
  float s = 0.f;
  const float* p = mu + (size_t)b * NN * EMB + e;
  for (int n = 0; n < NN; ++n) s += p[(size_t)n * EMB];
  cs[e] = s;
  __syncthreads();
  float g = b6[e];
  for (int k = 0; k < EMB; ++k) g += cs[k] * W6[k * EMB + e];
  rg[e] = g > 0.f ? g : 0.f;
  __syncthreads();
  float gp = b51[e];
  for (int k = 0; k < EMB; ++k) gp += rg[k] * W51[k * EMB + e];
  gpart[b * EMB + e] = gp;
}

__global__ __launch_bounds__(64) void k_final(
    const float* __restrict__ mu,
    const float* __restrict__ W7, const float* __restrict__ b7,
    const float* __restrict__ W51,
    const float* __restrict__ gpart,
    const float* __restrict__ w52, const float* __restrict__ b52,
    const int* __restrict__ reach,
    float* __restrict__ out) {                 // [B][512]
  __shared__ float murow[EMB];
  __shared__ float rl[EMB];
  __shared__ float red[EMB];
  const int bn = blockIdx.x;
  const int b  = bn >> 9;
  const int e  = threadIdx.x;
  murow[e] = mu[(size_t)bn * EMB + e];
  __syncthreads();
  float loc = b7[e];
  for (int k = 0; k < EMB; ++k) loc += murow[k] * W7[k * EMB + e];
  rl[e] = loc > 0.f ? loc : 0.f;
  __syncthreads();
  float r2 = gpart[b * EMB + e];
  for (int k = 0; k < EMB; ++k) r2 += rl[k] * W51[(EMB + k) * EMB + e];
  red[e] = r2 * w52[e];
  __syncthreads();
  for (int s = 32; s > 0; s >>= 1) {
    if (e < s) red[e] += red[e + s];
    __syncthreads();
  }
  if (e == 0) {
    float v = red[0] + b52[0];
    out[bn] = reach[bn] ? v : __uint_as_float(0xff800000u);   // -inf mask
  }
}

extern "C" void kernel_launch(void* const* d_in, const int* in_sizes, int n_in,
                              void* d_out, int out_size, void* d_ws, size_t ws_size,
                              hipStream_t stream) {
  (void)in_sizes; (void)n_in; (void)out_size; (void)ws_size;
  const float* xv    = (const float*)d_in[0];
  const float* Ws    = (const float*)d_in[1];
  const int*   reach = (const int*)d_in[2];
  const float* W1a   = (const float*)d_in[3];
  const float* b1a   = (const float*)d_in[4];
  const float* W1b   = (const float*)d_in[5];
  const float* b1b   = (const float*)d_in[6];
  const float* W2    = (const float*)d_in[7];
  const float* b2    = (const float*)d_in[8];
  const float* W3    = (const float*)d_in[9];
  const float* b3    = (const float*)d_in[10];
  const float* w4    = (const float*)d_in[11];
  const float* b4    = (const float*)d_in[12];
  const float* W6    = (const float*)d_in[13];
  const float* b6    = (const float*)d_in[14];
  const float* W7    = (const float*)d_in[15];
  const float* b7    = (const float*)d_in[16];
  const float* W51   = (const float*)d_in[17];
  const float* b51   = (const float*)d_in[18];
  const float* w52   = (const float*)d_in[19];
  const float* b52   = (const float*)d_in[20];

  char* ws = (char*)d_ws;
  unsigned short* WsBf = (unsigned short*)(ws + 0);         //  4 MiB
  unsigned short* muTA = (unsigned short*)(ws + 4194304);   // 512 KiB
  unsigned short* muTB = (unsigned short*)(ws + 4718592);   // 512 KiB
  unsigned short* W2T  = (unsigned short*)(ws + 5242880);   //   8 KiB
  float*          s13  = (float*)(ws + 5251072);            //   1 MiB
  float*          muF  = (float*)(ws + 6299648);            //   1 MiB
  float*          gpart= (float*)(ws + 7348224);            //   2 KiB

  k_cvt_ws<<<2048, 256, 0, stream>>>(Ws, WsBf, BATCH * NN * NN);
  k_w2t<<<16, 256, 0, stream>>>(W2, W2T);
  k_zero16<<<512, 256, 0, stream>>>(muTA, BATCH * EMB * NN);
  k_s13<<<BATCH * NN, 64, 0, stream>>>(xv, Ws, W1a, b1a, W1b, b1b, b2, W3, b3, w4, b4, s13);

  unsigned short* src = muTA;
  unsigned short* dst = muTB;
  for (int t = 0; t < TITER; ++t) {
    k_iter<<<32, 128, 0, stream>>>(WsBf, src, W2T, s13, muF, dst);
    unsigned short* tmp = src; src = dst; dst = tmp;
  }

  k_batch<<<BATCH, 64, 0, stream>>>(muF, W6, b6, W51, b51, gpart);
  k_final<<<BATCH * NN, 64, 0, stream>>>(muF, W7, b7, W51, gpart, w52, b52, reach, (float*)d_out);
}
